// MultiHeadAttention_79508434583676
// MI455X (gfx1250) — matrix operand order repaired and checked by reading, not compiled, not run on hardware
//
#include <hip/hip_runtime.h>

// ---- problem sizes (fixed by the reference) ----
// B=4, N=2048, D=512, H=8, DK=64.  Flattened token rows: B*N=8192.

typedef __attribute__((ext_vector_type(16))) __bf16    v16bf;
typedef __attribute__((ext_vector_type(8)))  float     v8f;
typedef __attribute__((ext_vector_type(8)))  unsigned  v8u;
typedef __attribute__((ext_vector_type(4)))  unsigned  v4u;
typedef __attribute__((ext_vector_type(4)))  float     v4f;

#define DEV static __device__ __forceinline__

DEV unsigned pk_bf16(float a, float b) {
  __bf16 lo = (__bf16)a, hi = (__bf16)b;
  unsigned short ul = __builtin_bit_cast(unsigned short, lo);
  unsigned short uh = __builtin_bit_cast(unsigned short, hi);
  return (unsigned)ul | ((unsigned)uh << 16);
}
DEV unsigned short bf16_bits(float a) {
  __bf16 x = (__bf16)a;
  return __builtin_bit_cast(unsigned short, x);
}
DEV float rcpf(float x) { return __builtin_amdgcn_rcpf(x); }
DEV float exp2f_fast(float x) { return __builtin_amdgcn_exp2f(x); }

DEV v8f wmma_bf16(v16bf a, v16bf b, v8f c) {
  // D = A(16x32 bf16) * B(32x16 bf16) + C(16x16 f32)
  return __builtin_amdgcn_wmma_f32_16x16x32_bf16(false, a, false, b, (short)0, c,
                                                 false, false);
}

// ---------- fragment loaders (ISA 7.12.2 wave32 layouts) ----------
// A 16x32 bf16: lane L -> row M=L&15, K-half kh=(L>=16)*8.
//   VGPR 0..3 hold K = kh..kh+7, VGPR 4..7 hold K = kh+16..kh+23  => two b128 loads.
DEV v16bf load_a_bf16(const unsigned short* p, int ld, int row, int kBase, int lane) {
  const unsigned short* base =
      p + (size_t)(row + (lane & 15)) * ld + kBase + ((lane >> 4) << 3);
  v4u lo = *(const v4u*)(base);
  v4u hi = *(const v4u*)(base + 16);
  v8u u = {lo[0], lo[1], lo[2], lo[3], hi[0], hi[1], hi[2], hi[3]};
  return __builtin_bit_cast(v16bf, u);
}

DEV v16bf load_a_f32(const float* p, int ld, int row, int kBase, int lane) {
  const float* base =
      p + (size_t)(row + (lane & 15)) * ld + kBase + ((lane >> 4) << 3);
  v4f f0 = *(const v4f*)(base);       // K kh+0..3
  v4f f1 = *(const v4f*)(base + 4);   // K kh+4..7
  v4f f2 = *(const v4f*)(base + 16);  // K kh+16..19
  v4f f3 = *(const v4f*)(base + 20);  // K kh+20..23
  v8u u = {pk_bf16(f0[0], f0[1]), pk_bf16(f0[2], f0[3]),
           pk_bf16(f1[0], f1[1]), pk_bf16(f1[2], f1[3]),
           pk_bf16(f2[0], f2[1]), pk_bf16(f2[2], f2[3]),
           pk_bf16(f3[0], f3[1]), pk_bf16(f3[2], f3[3])};
  return __builtin_bit_cast(v16bf, u);
}

// B 32x16 bf16: lane L -> col N=L&15, kh=(L>=16)*8; VGPR 0..3 hold K=kh..kh+7, VGPR 4..7 hold K=kh+16..kh+23.
// Source "col-major with K contiguous": element B[k][n] = src[n*ld + k].
DEV v16bf load_b_kcont_bf16(const unsigned short* p, int ld, int colBase, int kBase,
                            int lane) {
  const unsigned short* base =
      p + (size_t)(colBase + (lane & 15)) * ld + kBase + ((lane >> 4) << 3);
  v4u lo = *(const v4u*)(base);
  v4u hi = *(const v4u*)(base + 16);
  v8u u = {lo[0], lo[1], lo[2], lo[3], hi[0], hi[1], hi[2], hi[3]};
  return __builtin_bit_cast(v16bf, u);
}

DEV v16bf load_b_kcont_f32(const float* p, int ld, int colBase, int kBase, int lane) {
  const float* base =
      p + (size_t)(colBase + (lane & 15)) * ld + kBase + ((lane >> 4) << 3);
  v4f f0 = *(const v4f*)(base);
  v4f f1 = *(const v4f*)(base + 4);
  v4f f2 = *(const v4f*)(base + 16);
  v4f f3 = *(const v4f*)(base + 20);
  v8u u = {pk_bf16(f0[0], f0[1]), pk_bf16(f0[2], f0[3]),
           pk_bf16(f1[0], f1[1]), pk_bf16(f1[2], f1[3]),
           pk_bf16(f2[0], f2[1]), pk_bf16(f2[2], f2[3]),
           pk_bf16(f3[0], f3[1]), pk_bf16(f3[2], f3[3])};
  return __builtin_bit_cast(v16bf, u);
}

// C/D 16x16 f32: VGPR i, lane L -> M = i + 8*(L>=16), N = L&15
DEV void store_c_bias_scale_bf16(unsigned short* p, int ld, int row, int col, int lane,
                                 v8f c, const float* bias, float scale) {
  int cc = col + (lane & 15);
  int r0 = row + ((lane >> 4) << 3);
  float bv = bias[cc];
#pragma unroll
  for (int i = 0; i < 8; ++i) p[(size_t)(r0 + i) * ld + cc] = bf16_bits((c[i] + bv) * scale);
}

// Transposed store for V -> vpT[b][d][m]; 8 consecutive m values = one b128 store.
DEV void store_c_bias_bf16_T(unsigned short* p, int row, int col, int lane, v8f c,
                             const float* bias) {
  int cc = col + (lane & 15);
  int r0 = row + ((lane >> 4) << 3);
  int b = r0 >> 11, m = r0 & 2047;
  unsigned short* dst = p + ((size_t)b * 512 + cc) * 2048 + m;
  float bv = bias[cc];
  v4u u = {pk_bf16(c[0] + bv, c[1] + bv), pk_bf16(c[2] + bv, c[3] + bv),
           pk_bf16(c[4] + bv, c[5] + bv), pk_bf16(c[6] + bv, c[7] + bv)};
  *(v4u*)dst = u;
}

// =====================================================================
// Kernel 1: fused projection of Q, K, V with the shared Wq/bq (source bug).
// 24576 x 512 GEMM, K=512.  qp is pre-scaled by 0.125*log2(e) so the
// attention can use raw exp2 for the head-softmax.
// grid(384, 2), block 256 (8 waves).  Wave -> 32x64 output tile.
// =====================================================================
__global__ __launch_bounds__(256) void proj_qkv_kernel(
    const float* __restrict__ Q, const float* __restrict__ K, const float* __restrict__ V,
    const float* __restrict__ Wq, const float* __restrict__ bq,
    unsigned short* __restrict__ qp, unsigned short* __restrict__ kp,
    unsigned short* __restrict__ vpT) {
  const int lane = threadIdx.x & 31;
  const int wave = threadIdx.x >> 5;
  const int rowTile = blockIdx.x * 64 + (wave & 1) * 32;   // [0, 24576)
  const int colTile = blockIdx.y * 256 + (wave >> 1) * 64; // [0, 512)
  const int matId = rowTile >> 13;                         // 0:Q 1:K 2:V
  const int lr = rowTile & 8191;
  const float* X = (matId == 0) ? Q : ((matId == 1) ? K : V);

  v8f c[2][4];
#pragma unroll
  for (int ri = 0; ri < 2; ++ri)
#pragma unroll
    for (int ci = 0; ci < 4; ++ci) c[ri][ci] = {};

  for (int k0 = 0; k0 < 512; k0 += 32) {
    v16bf a[2], b[4];
#pragma unroll
    for (int ri = 0; ri < 2; ++ri) a[ri] = load_a_f32(X, 512, lr + ri * 16, k0, lane);
#pragma unroll
    for (int ci = 0; ci < 4; ++ci)
      b[ci] = load_b_kcont_f32(Wq, 512, colTile + ci * 16, k0, lane);
#pragma unroll
    for (int ri = 0; ri < 2; ++ri)
#pragma unroll
      for (int ci = 0; ci < 4; ++ci) c[ri][ci] = wmma_bf16(a[ri], b[ci], c[ri][ci]);
  }

  if (matId == 2) {
#pragma unroll
    for (int ri = 0; ri < 2; ++ri)
#pragma unroll
      for (int ci = 0; ci < 4; ++ci)
        store_c_bias_bf16_T(vpT, lr + ri * 16, colTile + ci * 16, lane, c[ri][ci], bq);
  } else {
    unsigned short* Dst = matId ? kp : qp;
    // 1/sqrt(64) * log2(e): scores land directly in the exp2 domain.
    const float scale = matId ? 1.0f : (0.125f * 1.44269504088896f);
#pragma unroll
    for (int ri = 0; ri < 2; ++ri)
#pragma unroll
      for (int ci = 0; ci < 4; ++ci)
        store_c_bias_scale_bf16(Dst, 512, lr + ri * 16, colTile + ci * 16, lane,
                                c[ri][ci], bq, scale);
  }
}

// =====================================================================
// Kernel 2: attention (softmax across HEADS, per the source bug) FUSED with
// the output projection.  grid(64, 4): x = 32-row q tile, y = batch.
// 8 waves = 8 heads.  LDS exchange is double-buffered: only 2 collective
// barriers per 32-wide m-step instead of 3.
// =====================================================================
__global__ __launch_bounds__(256) void attn_out_kernel(
    const unsigned short* __restrict__ qp, const unsigned short* __restrict__ kp,
    const unsigned short* __restrict__ vpT, const float* __restrict__ Wo,
    const float* __restrict__ bo, float* __restrict__ out) {
  __shared__ float rcpBuf[2][1024];   // ping-pong: rcp of cross-head sum
  __shared__ float eBuf[2][8][1024];  // ping-pong: per-head exp2(scores); 64 KB

  const int lane = threadIdx.x & 31;
  const int h = threadIdx.x >> 5;  // wave == head
  const int b = blockIdx.y;
  const int qRow = blockIdx.x * 32;
  const unsigned short* qpb = qp + (size_t)b * 2048 * 512;
  const unsigned short* kpb = kp + (size_t)b * 2048 * 512;
  const unsigned short* vtb = vpT + (size_t)b * 512 * 2048;

  // Q fragments are m-loop invariant (2 row tiles x 2 K chunks of 32).
  v16bf aq[2][2];
#pragma unroll
  for (int ri = 0; ri < 2; ++ri)
#pragma unroll
    for (int kc = 0; kc < 2; ++kc)
      aq[ri][kc] = load_a_bf16(qpb, 512, qRow + ri * 16, h * 64 + kc * 32, lane);

  v8f o[2][4];
#pragma unroll
  for (int ri = 0; ri < 2; ++ri)
#pragma unroll
    for (int ci = 0; ci < 4; ++ci) o[ri][ci] = {};

  for (int m0 = 0; m0 < 2048; m0 += 32) {
    const int pb = (m0 >> 5) & 1;  // ping-pong buffer select
    if (m0 + 32 < 2048) {
      __builtin_prefetch(kpb + (size_t)(m0 + 32) * 512 + h * 64, 0, 3);
      __builtin_prefetch(vtb + (size_t)(h * 64) * 2048 + (m0 + 32), 0, 3);
    }
    // ---- scores (already in exp2 domain via qp pre-scale), 32x32 tile ----
    v16bf bk[2][2];
#pragma unroll
    for (int mi = 0; mi < 2; ++mi)
#pragma unroll
      for (int kc = 0; kc < 2; ++kc)
        bk[mi][kc] = load_b_kcont_bf16(kpb, 512, m0 + mi * 16, h * 64 + kc * 32, lane);

    v8f e[2][2];
#pragma unroll
    for (int ri = 0; ri < 2; ++ri)
#pragma unroll
      for (int mi = 0; mi < 2; ++mi) {
        v8f s = {};
        s = wmma_bf16(aq[ri][0], bk[mi][0], s);
        s = wmma_bf16(aq[ri][1], bk[mi][1], s);
#pragma unroll
        for (int i = 0; i < 8; ++i) e[ri][mi][i] = exp2f_fast(s[i]);
      }

    // ---- publish exp tiles; deterministic cross-head sum -> reciprocal ----
#pragma unroll
    for (int ri = 0; ri < 2; ++ri)
#pragma unroll
      for (int mi = 0; mi < 2; ++mi)
#pragma unroll
        for (int i = 0; i < 8; ++i) {
          int idx = (ri * 16 + i + ((lane >> 4) << 3)) * 32 + mi * 16 + (lane & 15);
          eBuf[pb][h][idx] = e[ri][mi][i];
        }
    __syncthreads();
    {
      int idx = threadIdx.x * 4;  // 1024 entries / 256 threads, float4 each
      v4f acc = *(const v4f*)&eBuf[pb][0][idx];
#pragma unroll
      for (int hh = 1; hh < 8; ++hh) acc += *(const v4f*)&eBuf[pb][hh][idx];
      v4f r = {rcpf(acc[0]), rcpf(acc[1]), rcpf(acc[2]), rcpf(acc[3])};
      *(v4f*)&rcpBuf[pb][idx] = r;  // store 1/sum once; 8 head-waves reuse it
    }
    __syncthreads();

    // ---- P = e * rcp(sum), re-read in A-matrix layout (LDS = transpose) ----
    // No trailing barrier needed: next iteration writes the other buffer, and
    // its first barrier orders those writes after this iteration's reads.
    v16bf pa[2];
#pragma unroll
    for (int ri = 0; ri < 2; ++ri) {
      int rbase = (ri * 16 + (lane & 15)) * 32 + ((lane >> 4) << 3);
      const float* eb = &eBuf[pb][h][0];
      const float* rb = &rcpBuf[pb][0];
      v4f e0 = *(const v4f*)(eb + rbase);
      v4f e1 = *(const v4f*)(eb + rbase + 4);
      v4f e2 = *(const v4f*)(eb + rbase + 16);
      v4f e3 = *(const v4f*)(eb + rbase + 20);
      v4f r0 = *(const v4f*)(rb + rbase);
      v4f r1 = *(const v4f*)(rb + rbase + 4);
      v4f r2 = *(const v4f*)(rb + rbase + 16);
      v4f r3 = *(const v4f*)(rb + rbase + 20);
      v8u u = {pk_bf16(e0[0] * r0[0], e0[1] * r0[1]),
               pk_bf16(e0[2] * r0[2], e0[3] * r0[3]),
               pk_bf16(e1[0] * r1[0], e1[1] * r1[1]),
               pk_bf16(e1[2] * r1[2], e1[3] * r1[3]),
               pk_bf16(e2[0] * r2[0], e2[1] * r2[1]),
               pk_bf16(e2[2] * r2[2], e2[3] * r2[3]),
               pk_bf16(e3[0] * r3[0], e3[1] * r3[1]),
               pk_bf16(e3[2] * r3[2], e3[3] * r3[3])};
      pa[ri] = __builtin_bit_cast(v16bf, u);
    }

    // ---- O += P(32x32) * V(32x64); V from transposed vpT -> b128 loads ----
    v16bf bv[4];
#pragma unroll
    for (int ci = 0; ci < 4; ++ci)
      bv[ci] = load_b_kcont_bf16(vtb, 2048, h * 64 + ci * 16, m0, lane);
#pragma unroll
    for (int ri = 0; ri < 2; ++ri)
#pragma unroll
      for (int ci = 0; ci < 4; ++ci) o[ri][ci] = wmma_bf16(pa[ri], bv[ci], o[ri][ci]);
  }

  // =========== fused output projection: A(32x512, in LDS) @ Wo^T + bo ==========
  __syncthreads();  // all m-loop LDS reads complete before repurposing eBuf[0]
  unsigned short* aT = (unsigned short*)&eBuf[0][0][0];  // 32 KB bf16 tile [32][512]
#pragma unroll
  for (int ri = 0; ri < 2; ++ri)
#pragma unroll
    for (int ci = 0; ci < 4; ++ci) {
      int cc = h * 64 + ci * 16 + (lane & 15);
      int r0 = ri * 16 + ((lane >> 4) << 3);
#pragma unroll
      for (int i = 0; i < 8; ++i) aT[(r0 + i) * 512 + cc] = bf16_bits(o[ri][ci][i]);
    }
  __syncthreads();

  // Each wave computes out rows [qRow, qRow+32) x cols [wave*64, wave*64+64).
  v8f c2[2][4];
#pragma unroll
  for (int ri = 0; ri < 2; ++ri)
#pragma unroll
    for (int ci = 0; ci < 4; ++ci) c2[ri][ci] = {};

  for (int k0 = 0; k0 < 512; k0 += 32) {
    v16bf a2[2], b2[4];
#pragma unroll
    for (int ri = 0; ri < 2; ++ri) {
      const unsigned short* base =
          aT + (size_t)(ri * 16 + (lane & 15)) * 512 + k0 + ((lane >> 4) << 3);
      v4u lo = *(const v4u*)(base);
      v4u hi = *(const v4u*)(base + 16);
      v8u u = {lo[0], lo[1], lo[2], lo[3], hi[0], hi[1], hi[2], hi[3]};
      a2[ri] = __builtin_bit_cast(v16bf, u);
    }
#pragma unroll
    for (int ci = 0; ci < 4; ++ci)
      b2[ci] = load_b_kcont_f32(Wo, 512, h * 64 + ci * 16, k0, lane);
#pragma unroll
    for (int ri = 0; ri < 2; ++ri)
#pragma unroll
      for (int ci = 0; ci < 4; ++ci) c2[ri][ci] = wmma_bf16(a2[ri], b2[ci], c2[ri][ci]);
  }

  float* outb = out + (size_t)b * 2048 * 512;
#pragma unroll
  for (int ri = 0; ri < 2; ++ri)
#pragma unroll
    for (int ci = 0; ci < 4; ++ci) {
      int cc = h * 64 + ci * 16 + (lane & 15);
      int r0 = qRow + ri * 16 + ((lane >> 4) << 3);
      float bv = bo[cc];
#pragma unroll
      for (int i = 0; i < 8; ++i) outb[(size_t)(r0 + i) * 512 + cc] = c2[ri][ci][i] + bv;
    }
}

// =====================================================================
extern "C" void kernel_launch(void* const* d_in, const int* in_sizes, int n_in,
                              void* d_out, int out_size, void* d_ws, size_t ws_size,
                              hipStream_t stream) {
  (void)in_sizes; (void)n_in; (void)out_size; (void)ws_size;
  const float* Q  = (const float*)d_in[0];
  const float* K  = (const float*)d_in[1];
  const float* V  = (const float*)d_in[2];
  const float* Wq = (const float*)d_in[3];
  const float* bq = (const float*)d_in[4];
  const float* Wo = (const float*)d_in[5];
  const float* bo = (const float*)d_in[6];
  float* out = (float*)d_out;

  const size_t elems = (size_t)4 * 2048 * 512;  // 4 Mi elements per buffer
  unsigned short* qp  = (unsigned short*)d_ws;  // bf16 buffers in workspace
  unsigned short* kp  = qp + elems;
  unsigned short* vpT = kp + elems;             // [B][512][2048] transposed

  proj_qkv_kernel<<<dim3(384, 2), 256, 0, stream>>>(Q, K, V, Wq, bq, qp, kp, vpT);
  attn_out_kernel<<<dim3(64, 4), 256, 0, stream>>>(qp, kp, vpT, Wo, bo, out);
}
